// ScenePriorsGATModel_72670846648727
// MI455X (gfx1250) — matrix operand order repaired and checked by reading, not compiled, not run on hardware
//
#include <hip/hip_runtime.h>
#include <hip/hip_bf16.h>

// ---------------------------------------------------------------------------
// ScenePriors GAT model for MI455X (gfx1250, wave32, WMMA).
// Heavy matmuls (h@W [4096x1024x1024], P@Wh [4096x4096x1024]) run on
// v_wmma_f32_16x16x32_f16 with f32 accumulate, 32x64 tile per wave.
// K-loop unrolled x64 with explicit ping-pong fragment buffers: no register
// rotation movs, immediate-offset reloads, one pointer bump per stream.
// ---------------------------------------------------------------------------

typedef _Float16 v16h __attribute__((ext_vector_type(16)));
typedef _Float16 h8   __attribute__((ext_vector_type(8)));
typedef float    v8f  __attribute__((ext_vector_type(8)));

#define NNODE 4096
#define FDIM  1024
#define BBATCH 4
#define GALPHA 0.02f
#define NEGV  (-9.0e15f)

// ---- workspace layout (bytes) ---------------------------------------------
#define OFF_HF16  ((size_t)0)                       // [4096*1024] f16   8 MB
#define OFF_WT16  (OFF_HF16 + (size_t)NNODE*FDIM*2) // [1024*1024] f16   2 MB
#define OFF_WH    (OFF_WT16 + (size_t)FDIM*FDIM*2)  // [4096*1024] f32  16 MB
#define OFF_WHT   (OFF_WH   + (size_t)NNODE*FDIM*4) // [1024*4096] f16   8 MB
#define OFF_P     (OFF_WHT  + (size_t)FDIM*NNODE*2) // [4096*4096] f16  32 MB
#define OFF_HOUT  (OFF_P    + (size_t)NNODE*NNODE*2)// [4096*1024] f32  16 MB
#define OFF_SSRC  (OFF_HOUT + (size_t)NNODE*FDIM*4)
#define OFF_SDST  (OFF_SSRC + (size_t)NNODE*4)
#define OFF_RMAX  (OFF_SDST + (size_t)NNODE*4)
#define OFF_RSUM  (OFF_RMAX + (size_t)NNODE*4)
#define OFF_X     (OFF_RSUM + (size_t)NNODE*4)      // [4,512]
#define OFF_Y     (OFF_X    + (size_t)BBATCH*512*4) // [4,300]
#define OFF_RES   (OFF_Y    + (size_t)BBATCH*300*4) // [512]
#define OFF_WH3   (OFF_RES  + (size_t)512*4)        // [4096]
#define OFF_HCOL  (OFF_WH3  + (size_t)NNODE*4)      // [4096]
#define OFF_Z     (OFF_HCOL + (size_t)NNODE*4)      // [512]
#define OFF_U     (OFF_Z    + (size_t)512*4)        // [4,512]
#define OFF_V2    (OFF_U    + (size_t)BBATCH*512*4) // [4,512]

__device__ __forceinline__ float waveSum(float v) {
#pragma unroll
  for (int o = 16; o > 0; o >>= 1) v += __shfl_down(v, o, 32);
  return v;
}

// A 16x32 f16 fragment: per-lane chunks at +0 and +16 halves (half-wave K-interleave)
__device__ __forceinline__ v16h loadFragA(const _Float16* __restrict__ p) {
  h8 c0 = *(const h8*)(p);
  h8 c1 = *(const h8*)(p + 16);
  v16h f;
#pragma unroll
  for (int e = 0; e < 8; ++e) { f[e] = c0[e]; f[8 + e] = c1[e]; }
  return f;
}

// B 32x16 f16 fragment from Bt[N,K]: per-lane chunks at +0 and +16 halves (half-wave K-interleave)
__device__ __forceinline__ v16h loadFragB(const _Float16* __restrict__ p) {
  h8 c0 = *(const h8*)(p);
  h8 c1 = *(const h8*)(p + 16);
  v16h f;
#pragma unroll
  for (int e = 0; e < 8; ++e) { f[e] = c0[e]; f[8 + e] = c1[e]; }
  return f;
}

// ---------------------------------------------------------------------------
// Generic wave-per-output linear: out[r,o] = act(in[r,:] . w[o,:] + b[o])
// ---------------------------------------------------------------------------
__global__ void k_linear_rows(const float* __restrict__ in, const float* __restrict__ w,
                              const float* __restrict__ b, float* __restrict__ out,
                              int rows, int I, int O, int relu) {
  int wid  = (blockIdx.x * blockDim.x + threadIdx.x) >> 5;
  int lane = threadIdx.x & 31;
  if (wid >= rows * O) return;
  int r = wid / O, o = wid % O;
  const float* ip = in + (size_t)r * I;
  const float* wp = w  + (size_t)o * I;
  float s = 0.f;
  for (int c = lane; c < I; c += 32) s += ip[c] * wp[c];
  s = waveSum(s);
  if (lane == 0) {
    s += b[o];
    if (relu) s = fmaxf(s, 0.f);
    out[(size_t)r * O + o] = s;
  }
}

// ---------------------------------------------------------------------------
// h = [broadcast(res,512) | all_glove @ word_w^T + word_b]  -> f16
// ---------------------------------------------------------------------------
__global__ void k_build_h(const float* __restrict__ res, const float* __restrict__ all_glove,
                          const float* __restrict__ word_w, const float* __restrict__ word_b,
                          _Float16* __restrict__ hf16) {
  size_t stride = (size_t)gridDim.x * blockDim.x;
  for (size_t idx = (size_t)blockIdx.x * blockDim.x + threadIdx.x;
       idx < (size_t)NNODE * FDIM; idx += stride) {
    int i = (int)(idx >> 10);     // FDIM = 1024
    int c = (int)(idx & 1023);
    float v;
    if (c < 512) {
      v = res[c];
    } else {
      int o = c - 512;
      const float* g  = all_glove + (size_t)i * 300;
      const float* wp = word_w + (size_t)o * 300;
      float s = word_b[o];
      for (int k = 0; k < 300; ++k) s += g[k] * wp[k];
      v = s;
    }
    hf16[idx] = (_Float16)v;
  }
}

// dst[c*R + r] = (f16) src[r*C + c]
__global__ void k_transpose_f32_to_f16(const float* __restrict__ src,
                                       _Float16* __restrict__ dst, int R, int C) {
  size_t total = (size_t)R * C;
  size_t stride = (size_t)gridDim.x * blockDim.x;
  for (size_t idx = (size_t)blockIdx.x * blockDim.x + threadIdx.x; idx < total; idx += stride) {
    int r = (int)(idx / C), c = (int)(idx % C);
    dst[(size_t)c * R + r] = (_Float16)src[idx];
  }
}

__global__ void k_f32_to_f16(const float* __restrict__ src, _Float16* __restrict__ dst, size_t n) {
  size_t stride = (size_t)gridDim.x * blockDim.x;
  for (size_t idx = (size_t)blockIdx.x * blockDim.x + threadIdx.x; idx < n; idx += stride)
    dst[idx] = (_Float16)src[idx];
}

// ---------------------------------------------------------------------------
// WMMA GEMM: C[M,Nn](f32) = A[M,K](f16, row-major) x Bt[Nn,K](f16, row-major)^T
// One wave owns a 32x64 tile: 2 A-frags x 4 B-frags -> 8 WMMA per 32-K step.
// K unrolled x64 with ping-pong buffers S0/S1 kept in fixed registers.
// Requires K % 64 == 0 (true for 1024 and 4096).
// ---------------------------------------------------------------------------
__global__ __launch_bounds__(128)
void k_wmma_gemm(const _Float16* __restrict__ A, const _Float16* __restrict__ Bt,
                 float* __restrict__ C, const float* __restrict__ bias,
                 int M, int K, int Nn, int relu, int totalWT) {
  const int lane = threadIdx.x & 31;
  const int wave = threadIdx.x >> 5;
  const int wt   = blockIdx.x * 4 + wave;
  if (wt >= totalWT) return;                 // uniform per wave
  const int tilesN4 = Nn >> 6;
  const int tM2 = wt / tilesN4;              // 32-row block
  const int tN4 = wt % tilesN4;              // 64-col block
  const int half = lane >> 4;                // K-interleave half-wave select
  const int l16  = lane & 15;

  const _Float16* Ap0 = A + (size_t)(tM2 * 32 + l16)      * K + 8 * half;
  const _Float16* Ap1 = A + (size_t)(tM2 * 32 + 16 + l16) * K + 8 * half;
  const _Float16* Bp0 = Bt + (size_t)((tN4 * 4 + 0) * 16 + l16) * K + 8 * half;
  const _Float16* Bp1 = Bt + (size_t)((tN4 * 4 + 1) * 16 + l16) * K + 8 * half;
  const _Float16* Bp2 = Bt + (size_t)((tN4 * 4 + 2) * 16 + l16) * K + 8 * half;
  const _Float16* Bp3 = Bt + (size_t)((tN4 * 4 + 3) * 16 + l16) * K + 8 * half;

  v8f acc[2][4];
#pragma unroll
  for (int m = 0; m < 2; ++m)
#pragma unroll
    for (int t = 0; t < 4; ++t)
#pragma unroll
      for (int v = 0; v < 8; ++v) acc[m][t][v] = 0.f;

  // prologue: S0 @ kk=0, S1 @ kk=32
  v16h a0_0 = loadFragA(Ap0);      v16h a1_0 = loadFragA(Ap1);
  v16h b0_0 = loadFragB(Bp0);      v16h b1_0 = loadFragB(Bp1);
  v16h b2_0 = loadFragB(Bp2);      v16h b3_0 = loadFragB(Bp3);
  v16h a0_1 = loadFragA(Ap0 + 32); v16h a1_1 = loadFragA(Ap1 + 32);
  v16h b0_1 = loadFragB(Bp0 + 32); v16h b1_1 = loadFragB(Bp1 + 32);
  v16h b2_1 = loadFragB(Bp2 + 32); v16h b3_1 = loadFragB(Bp3 + 32);

  const int iters = (K >> 6) - 1;  // K/64 - 1 steady-state iterations
  for (int it = 0; it < iters; ++it) {
    // compute S0, then reload S0 from +64 halves (immediate offsets)
    acc[0][0] = __builtin_amdgcn_wmma_f32_16x16x32_f16(false, a0_0, false, b0_0, (short)0, acc[0][0], false, false);
    acc[1][0] = __builtin_amdgcn_wmma_f32_16x16x32_f16(false, a1_0, false, b0_0, (short)0, acc[1][0], false, false);
    acc[0][1] = __builtin_amdgcn_wmma_f32_16x16x32_f16(false, a0_0, false, b1_0, (short)0, acc[0][1], false, false);
    acc[1][1] = __builtin_amdgcn_wmma_f32_16x16x32_f16(false, a1_0, false, b1_0, (short)0, acc[1][1], false, false);
    acc[0][2] = __builtin_amdgcn_wmma_f32_16x16x32_f16(false, a0_0, false, b2_0, (short)0, acc[0][2], false, false);
    acc[1][2] = __builtin_amdgcn_wmma_f32_16x16x32_f16(false, a1_0, false, b2_0, (short)0, acc[1][2], false, false);
    acc[0][3] = __builtin_amdgcn_wmma_f32_16x16x32_f16(false, a0_0, false, b3_0, (short)0, acc[0][3], false, false);
    acc[1][3] = __builtin_amdgcn_wmma_f32_16x16x32_f16(false, a1_0, false, b3_0, (short)0, acc[1][3], false, false);
    a0_0 = loadFragA(Ap0 + 64); a1_0 = loadFragA(Ap1 + 64);
    b0_0 = loadFragB(Bp0 + 64); b1_0 = loadFragB(Bp1 + 64);
    b2_0 = loadFragB(Bp2 + 64); b3_0 = loadFragB(Bp3 + 64);

    // compute S1, then reload S1 from +96 halves (immediate offsets)
    acc[0][0] = __builtin_amdgcn_wmma_f32_16x16x32_f16(false, a0_1, false, b0_1, (short)0, acc[0][0], false, false);
    acc[1][0] = __builtin_amdgcn_wmma_f32_16x16x32_f16(false, a1_1, false, b0_1, (short)0, acc[1][0], false, false);
    acc[0][1] = __builtin_amdgcn_wmma_f32_16x16x32_f16(false, a0_1, false, b1_1, (short)0, acc[0][1], false, false);
    acc[1][1] = __builtin_amdgcn_wmma_f32_16x16x32_f16(false, a1_1, false, b1_1, (short)0, acc[1][1], false, false);
    acc[0][2] = __builtin_amdgcn_wmma_f32_16x16x32_f16(false, a0_1, false, b2_1, (short)0, acc[0][2], false, false);
    acc[1][2] = __builtin_amdgcn_wmma_f32_16x16x32_f16(false, a1_1, false, b2_1, (short)0, acc[1][2], false, false);
    acc[0][3] = __builtin_amdgcn_wmma_f32_16x16x32_f16(false, a0_1, false, b3_1, (short)0, acc[0][3], false, false);
    acc[1][3] = __builtin_amdgcn_wmma_f32_16x16x32_f16(false, a1_1, false, b3_1, (short)0, acc[1][3], false, false);
    a0_1 = loadFragA(Ap0 + 96); a1_1 = loadFragA(Ap1 + 96);
    b0_1 = loadFragB(Bp0 + 96); b1_1 = loadFragB(Bp1 + 96);
    b2_1 = loadFragB(Bp2 + 96); b3_1 = loadFragB(Bp3 + 96);

    // one pointer bump per stream per 64-K iteration
    Ap0 += 64; Ap1 += 64; Bp0 += 64; Bp1 += 64; Bp2 += 64; Bp3 += 64;
  }

  // epilogue: last two 32-K steps
  acc[0][0] = __builtin_amdgcn_wmma_f32_16x16x32_f16(false, a0_0, false, b0_0, (short)0, acc[0][0], false, false);
  acc[1][0] = __builtin_amdgcn_wmma_f32_16x16x32_f16(false, a1_0, false, b0_0, (short)0, acc[1][0], false, false);
  acc[0][1] = __builtin_amdgcn_wmma_f32_16x16x32_f16(false, a0_0, false, b1_0, (short)0, acc[0][1], false, false);
  acc[1][1] = __builtin_amdgcn_wmma_f32_16x16x32_f16(false, a1_0, false, b1_0, (short)0, acc[1][1], false, false);
  acc[0][2] = __builtin_amdgcn_wmma_f32_16x16x32_f16(false, a0_0, false, b2_0, (short)0, acc[0][2], false, false);
  acc[1][2] = __builtin_amdgcn_wmma_f32_16x16x32_f16(false, a1_0, false, b2_0, (short)0, acc[1][2], false, false);
  acc[0][3] = __builtin_amdgcn_wmma_f32_16x16x32_f16(false, a0_0, false, b3_0, (short)0, acc[0][3], false, false);
  acc[1][3] = __builtin_amdgcn_wmma_f32_16x16x32_f16(false, a1_0, false, b3_0, (short)0, acc[1][3], false, false);

  acc[0][0] = __builtin_amdgcn_wmma_f32_16x16x32_f16(false, a0_1, false, b0_1, (short)0, acc[0][0], false, false);
  acc[1][0] = __builtin_amdgcn_wmma_f32_16x16x32_f16(false, a1_1, false, b0_1, (short)0, acc[1][0], false, false);
  acc[0][1] = __builtin_amdgcn_wmma_f32_16x16x32_f16(false, a0_1, false, b1_1, (short)0, acc[0][1], false, false);
  acc[1][1] = __builtin_amdgcn_wmma_f32_16x16x32_f16(false, a1_1, false, b1_1, (short)0, acc[1][1], false, false);
  acc[0][2] = __builtin_amdgcn_wmma_f32_16x16x32_f16(false, a0_1, false, b2_1, (short)0, acc[0][2], false, false);
  acc[1][2] = __builtin_amdgcn_wmma_f32_16x16x32_f16(false, a1_1, false, b2_1, (short)0, acc[1][2], false, false);
  acc[0][3] = __builtin_amdgcn_wmma_f32_16x16x32_f16(false, a0_1, false, b3_1, (short)0, acc[0][3], false, false);
  acc[1][3] = __builtin_amdgcn_wmma_f32_16x16x32_f16(false, a1_1, false, b3_1, (short)0, acc[1][3], false, false);

  // C/D layout: N = lane%16, M = vgpr + 8*(lane/16)
#pragma unroll
  for (int m = 0; m < 2; ++m) {
#pragma unroll
    for (int t = 0; t < 4; ++t) {
      const int col = (tN4 * 4 + t) * 16 + l16;
      float bv = bias ? bias[col] : 0.f;
#pragma unroll
      for (int v = 0; v < 8; ++v) {
        int row = tM2 * 32 + m * 16 + v + 8 * half;
        float x = acc[m][t][v] + bv;
        if (relu) x = fmaxf(x, 0.f);
        C[(size_t)row * Nn + col] = x;
      }
    }
  }
}

// ---------------------------------------------------------------------------
// s_src[i] = Wh[i,:] . A[0:Fo],  s_dst[i] = Wh[i,:] . A[Fo:2Fo]
// ---------------------------------------------------------------------------
__global__ void k_rowdots(const float* __restrict__ Wh, const float* __restrict__ Avec,
                          float* __restrict__ ssrc, float* __restrict__ sdst, int Fo) {
  int row  = (blockIdx.x * blockDim.x + threadIdx.x) >> 5;
  int lane = threadIdx.x & 31;
  if (row >= NNODE) return;
  const float* wp = Wh + (size_t)row * Fo;
  float s1 = 0.f, s2 = 0.f;
  for (int c = lane; c < Fo; c += 32) { float v = wp[c]; s1 += v * Avec[c]; s2 += v * Avec[Fo + c]; }
  s1 = waveSum(s1); s2 = waveSum(s2);
  if (lane == 0) { ssrc[row] = s1; sdst[row] = s2; }
}

// ---------------------------------------------------------------------------
// Online-softmax row stats over masked leaky-relu scores.
// ---------------------------------------------------------------------------
__global__ void k_softmax_stats(const float* __restrict__ ssrc, const float* __restrict__ sdst,
                                const float* __restrict__ adj,
                                float* __restrict__ rmax, float* __restrict__ rsum) {
  int row  = (blockIdx.x * blockDim.x + threadIdx.x) >> 5;
  int lane = threadIdx.x & 31;
  if (row >= NNODE) return;
  float si = ssrc[row];
  const float* arow = adj + (size_t)row * NNODE;
  float m = -INFINITY, s = 0.f;
  for (int j = lane; j < NNODE; j += 32) {
    float e = si + sdst[j];
    e = e > 0.f ? e : GALPHA * e;
    e = arow[j] > 0.f ? e : NEGV;
    if (e > m) { s = s * __expf(m - e) + 1.f; m = e; }
    else       { s += __expf(e - m); }
  }
#pragma unroll
  for (int o = 16; o > 0; o >>= 1) {
    float mo = __shfl_down(m, o, 32);
    float so = __shfl_down(s, o, 32);
    float mn = fmaxf(m, mo);
    s = s * __expf(m - mn) + so * __expf(mo - mn);
    m = mn;
  }
  if (lane == 0) { rmax[row] = m; rsum[row] = s; }
}

// P[i,j] = f16( adj>0 ? exp(lrelu(ssrc+sdst)-m_i)/sum_i : 0 )
__global__ void k_emit_P(const float* __restrict__ ssrc, const float* __restrict__ sdst,
                         const float* __restrict__ adj, const float* __restrict__ rmax,
                         const float* __restrict__ rsum, _Float16* __restrict__ P) {
  size_t total  = (size_t)NNODE * NNODE;
  size_t stride = (size_t)gridDim.x * blockDim.x;
  for (size_t idx = (size_t)blockIdx.x * blockDim.x + threadIdx.x; idx < total; idx += stride) {
    int i = (int)(idx >> 12);      // NNODE = 4096
    int j = (int)(idx & 4095);
    float p = 0.f;
    if (adj[idx] > 0.f) {
      float e = ssrc[i] + sdst[j];
      e = e > 0.f ? e : GALPHA * e;
      p = __expf(e - rmax[i]) / rsum[i];
    }
    P[idx] = (_Float16)p;
  }
}

// ---------------------------------------------------------------------------
// Layer 3 (Fo = 1): scalar GAT.
// ---------------------------------------------------------------------------
__global__ void k_wh3(const float* __restrict__ h, const float* __restrict__ W3,
                      const float* __restrict__ A3, float* __restrict__ wh3,
                      float* __restrict__ ssrc, float* __restrict__ sdst) {
  int row  = (blockIdx.x * blockDim.x + threadIdx.x) >> 5;
  int lane = threadIdx.x & 31;
  if (row >= NNODE) return;
  const float* hp = h + (size_t)row * FDIM;
  float s = 0.f;
  for (int c = lane; c < FDIM; c += 32) s += hp[c] * W3[c];
  s = waveSum(s);
  if (lane == 0) { wh3[row] = s; ssrc[row] = s * A3[0]; sdst[row] = s * A3[1]; }
}

__global__ void k_gat3(const float* __restrict__ ssrc, const float* __restrict__ sdst,
                       const float* __restrict__ adj, const float* __restrict__ rmax,
                       const float* __restrict__ rsum, const float* __restrict__ wh3,
                       const float* __restrict__ b3, float* __restrict__ hcol) {
  int row  = (blockIdx.x * blockDim.x + threadIdx.x) >> 5;
  int lane = threadIdx.x & 31;
  if (row >= NNODE) return;
  float si = ssrc[row], mi = rmax[row], inv = 1.f / rsum[row];
  const float* arow = adj + (size_t)row * NNODE;
  float acc = 0.f;
  for (int j = lane; j < NNODE; j += 32) {
    if (arow[j] > 0.f) {
      float e = si + sdst[j];
      e = e > 0.f ? e : GALPHA * e;
      acc += __expf(e - mi) * wh3[j];
    }
  }
  acc = waveSum(acc);
  if (lane == 0) hcol[row] = fmaxf(acc * inv + b3[0], 0.f);
}

// ---------------------------------------------------------------------------
// navi layer with fused xyz gather: input c -> x | y | z
// ---------------------------------------------------------------------------
__global__ void k_navi(const float* __restrict__ x, const float* __restrict__ y,
                       const float* __restrict__ z, const float* __restrict__ w,
                       const float* __restrict__ b, float* __restrict__ u) {
  int wid  = (blockIdx.x * blockDim.x + threadIdx.x) >> 5;
  int lane = threadIdx.x & 31;
  if (wid >= BBATCH * 512) return;
  int bb = wid / 512, o = wid % 512;
  const float* wp = w + (size_t)o * 940;
  float s = 0.f;
  for (int c = lane; c < 940; c += 32) {
    float iv = (c < 512) ? x[bb * 512 + c]
             : (c < 812) ? y[bb * 300 + (c - 512)]
                         : z[bb * 128 + (c - 812)];
    s += iv * wp[c];
  }
  s = waveSum(s);
  if (lane == 0) u[bb * 512 + o] = fmaxf(s + b[o], 0.f);
}

// policy (24) then value (2048) into d_out
__global__ void k_headout(const float* __restrict__ v2,
                          const float* __restrict__ actor_w, const float* __restrict__ actor_b,
                          const float* __restrict__ critic_w, const float* __restrict__ critic_b,
                          float* __restrict__ out) {
  int wid  = (blockIdx.x * blockDim.x + threadIdx.x) >> 5;
  int lane = threadIdx.x & 31;
  if (wid >= 24 + BBATCH * 512) return;
  const float* wp; const float* vp; float bv; int oidx;
  if (wid < 24) {
    int bb = wid / 6, o = wid % 6;
    wp = actor_w + (size_t)o * 512; vp = v2 + (size_t)bb * 512;
    bv = actor_b[o]; oidx = wid;
  } else {
    int t = wid - 24, bb = t / 512, o = t % 512;
    wp = critic_w + (size_t)o * 512; vp = v2 + (size_t)bb * 512;
    bv = critic_b[o]; oidx = 24 + t;
  }
  float s = 0.f;
  for (int c = lane; c < 512; c += 32) s += vp[c] * wp[c];
  s = waveSum(s);
  if (lane == 0) out[oidx] = s + bv;
}

// ---------------------------------------------------------------------------
extern "C" void kernel_launch(void* const* d_in, const int* in_sizes, int n_in,
                              void* d_out, int out_size, void* d_ws, size_t ws_size,
                              hipStream_t stream) {
  (void)in_sizes; (void)n_in; (void)out_size; (void)ws_size;
  const float* fc4         = (const float*)d_in[0];
  const float* glove       = (const float*)d_in[1];
  const float* score       = (const float*)d_in[2];
  const float* all_glove   = (const float*)d_in[3];
  const float* adj         = (const float*)d_in[4];
  const float* fc_state_w  = (const float*)d_in[5];
  const float* fc_state_b  = (const float*)d_in[6];
  const float* fc_target_w = (const float*)d_in[7];
  const float* fc_target_b = (const float*)d_in[8];
  const float* word_w      = (const float*)d_in[9];
  const float* word_b      = (const float*)d_in[10];
  const float* resnet_w    = (const float*)d_in[11];
  const float* resnet_b    = (const float*)d_in[12];
  const float* W1 = (const float*)d_in[13];
  const float* b1 = (const float*)d_in[14];
  const float* A1 = (const float*)d_in[15];
  const float* W2 = (const float*)d_in[16];
  const float* b2 = (const float*)d_in[17];
  const float* A2 = (const float*)d_in[18];
  const float* W3 = (const float*)d_in[19];
  const float* b3 = (const float*)d_in[20];
  const float* A3 = (const float*)d_in[21];
  const float* map_w  = (const float*)d_in[22];
  const float* map_b  = (const float*)d_in[23];
  const float* navi_w = (const float*)d_in[24];
  const float* navi_b = (const float*)d_in[25];
  const float* hid_w  = (const float*)d_in[26];
  const float* hid_b  = (const float*)d_in[27];
  const float* actor_w  = (const float*)d_in[28];
  const float* actor_b  = (const float*)d_in[29];
  const float* critic_w = (const float*)d_in[30];
  const float* critic_b = (const float*)d_in[31];
  float* out = (float*)d_out;

  char* w = (char*)d_ws;
  _Float16* hf16 = (_Float16*)(w + OFF_HF16);
  _Float16* Wt16 = (_Float16*)(w + OFF_WT16);
  float*    Wh   = (float*)   (w + OFF_WH);
  _Float16* WhT  = (_Float16*)(w + OFF_WHT);
  _Float16* P    = (_Float16*)(w + OFF_P);
  float*    hout = (float*)   (w + OFF_HOUT);
  float*    ssrc = (float*)   (w + OFF_SSRC);
  float*    sdst = (float*)   (w + OFF_SDST);
  float*    rmax = (float*)   (w + OFF_RMAX);
  float*    rsum = (float*)   (w + OFF_RSUM);
  float*    xb   = (float*)   (w + OFF_X);
  float*    yb   = (float*)   (w + OFF_Y);
  float*    resb = (float*)   (w + OFF_RES);
  float*    wh3  = (float*)   (w + OFF_WH3);
  float*    hcol = (float*)   (w + OFF_HCOL);
  float*    zb   = (float*)   (w + OFF_Z);
  float*    ub   = (float*)   (w + OFF_U);
  float*    v2b  = (float*)   (w + OFF_V2);

  const int TPB = 256, WPB = TPB / 32;
  auto wblocks = [&](int waves) { return (waves + WPB - 1) / WPB; };

  // --- input embeddings
  k_linear_rows<<<wblocks(BBATCH * 512), TPB, 0, stream>>>(fc4, fc_state_w, fc_state_b, xb, BBATCH, 8192, 512, 1);
  k_linear_rows<<<wblocks(BBATCH * 300), TPB, 0, stream>>>(glove, fc_target_w, fc_target_b, yb, BBATCH, 300, 300, 1);
  k_linear_rows<<<wblocks(512), TPB, 0, stream>>>(score, resnet_w, resnet_b, resb, 1, 1000, 512, 0);
  k_build_h<<<2048, TPB, 0, stream>>>(resb, all_glove, word_w, word_b, hf16);

  // --- GAT layers 1 and 2 (dense, WMMA)
  const int gemmWT = (NNODE / 32) * (FDIM / 64);       // 2048 wave-tiles (32x64 each)
  const int gemmBlocks = gemmWT / 4;
  const float* Ws[2] = { W1, W2 };
  const float* As[2] = { A1, A2 };
  const float* bs[2] = { b1, b2 };
  for (int layer = 0; layer < 2; ++layer) {
    k_transpose_f32_to_f16<<<2048, TPB, 0, stream>>>(Ws[layer], Wt16, FDIM, FDIM);
    k_wmma_gemm<<<gemmBlocks, 128, 0, stream>>>(hf16, Wt16, Wh, nullptr, NNODE, FDIM, FDIM, 0, gemmWT);
    k_rowdots<<<wblocks(NNODE), TPB, 0, stream>>>(Wh, As[layer], ssrc, sdst, FDIM);
    k_softmax_stats<<<wblocks(NNODE), TPB, 0, stream>>>(ssrc, sdst, adj, rmax, rsum);
    k_emit_P<<<4096, TPB, 0, stream>>>(ssrc, sdst, adj, rmax, rsum, P);
    k_transpose_f32_to_f16<<<4096, TPB, 0, stream>>>(Wh, WhT, NNODE, FDIM);
    k_wmma_gemm<<<gemmBlocks, 128, 0, stream>>>(P, WhT, hout, bs[layer], NNODE, NNODE, FDIM, 1, gemmWT);
    if (layer == 0)
      k_f32_to_f16<<<2048, TPB, 0, stream>>>(hout, hf16, (size_t)NNODE * FDIM);
  }

  // --- GAT layer 3 (Fo = 1)
  k_wh3<<<wblocks(NNODE), TPB, 0, stream>>>(hout, W3, A3, wh3, ssrc, sdst);
  k_softmax_stats<<<wblocks(NNODE), TPB, 0, stream>>>(ssrc, sdst, adj, rmax, rsum);
  k_gat3<<<wblocks(NNODE), TPB, 0, stream>>>(ssrc, sdst, adj, rmax, rsum, wh3, b3, hcol);

  // --- head
  k_linear_rows<<<wblocks(512), TPB, 0, stream>>>(hcol, map_w, map_b, zb, 1, NNODE, 512, 0);
  k_navi<<<wblocks(BBATCH * 512), TPB, 0, stream>>>(xb, yb, zb, navi_w, navi_b, ub);
  k_linear_rows<<<wblocks(BBATCH * 512), TPB, 0, stream>>>(ub, hid_w, hid_b, v2b, BBATCH, 512, 512, 1);
  k_headout<<<wblocks(24 + BBATCH * 512), TPB, 0, stream>>>(v2b, actor_w, actor_b, critic_w, critic_b, out);
}
